// DSGCNEncoder_10823317586643
// MI455X (gfx1250) — compile-verified, matrix operand order checked by reading, not run on hardware
//
#include <hip/hip_runtime.h>
#include <hip/hip_bf16.h>

// ---------------------------------------------------------------------------
// DSGCN encoder for MI455X (gfx1250, wave32).  All dense GEMMs run on
// v_wmma_f32_16x16x32_bf16 (f32 -> bf16 convert while staging to LDS,
// f32 accumulate).  Fragments are loaded as contiguous ds_load_b128 pairs
// by staging A row-major and B transposed in LDS.
// ---------------------------------------------------------------------------

#define BDIM 64
#define TDIM 32
#define NDIM 47
#define DDIM 256
#define M1   (BDIM * TDIM * NDIM)   // 96256 node rows  (divisible by 128)
#define M2   (BDIM * TDIM)          // 2048 token rows  (divisible by 128)

typedef __attribute__((ext_vector_type(16))) __bf16 v16bf;
typedef __attribute__((ext_vector_type(8)))  __bf16 v8bf;
typedef __attribute__((ext_vector_type(2)))  __bf16 v2bf;
typedef __attribute__((ext_vector_type(8)))  float  v8f;

#define BPAD 40   // padded transposed-B row length (80 bytes: 16B aligned)

__device__ __forceinline__ float gelu_f(float x) {
    return 0.5f * x * (1.0f + erff(x * 0.70710678118654752f));
}
__device__ __forceinline__ float wave_sum(float v) {
    #pragma unroll
    for (int o = 16; o >= 1; o >>= 1) v += __shfl_xor(v, o, 32);
    return v;
}
__device__ __forceinline__ float wave_max(float v) {
    #pragma unroll
    for (int o = 16; o >= 1; o >>= 1) v = fmaxf(v, __shfl_xor(v, o, 32));
    return v;
}

// ---------------------------------------------------------------------------
// WMMA GEMM:  C[M,N] = A[M,K] @ B[K,N]  (+bias) (+GELU), all row-major f32.
// Block tile 128x64, 8 waves (4x2), wave tile 32x32 = 2x2 wmma 16x16 tiles.
// AMODE==1: A is formed on the fly from X [M,Cx] and scale table S [47, K]:
//           A[row, kk] = X[row, kk % Cx] * S[row % 47, kk]   (GCN expand)
// GUARDK: zero-pad K tail (only geo GEMM, K=332).  M % 128 == 0, N % 64 == 0.
// ---------------------------------------------------------------------------
template<int AMODE, int EPI, bool GUARDK>
__global__ __launch_bounds__(256) void gemm_kernel(
    const float* __restrict__ A, const float* __restrict__ Bw,
    const float* __restrict__ bias, float* __restrict__ C,
    int M, int Nn, int K,
    const float* __restrict__ X, const float* __restrict__ S, int Cx)
{
    (void)M;
    __shared__ __bf16 As[128 * 32];       // row-major [m][k]
    __shared__ __bf16 Bst[64 * BPAD];     // transposed [n][k], padded

    const int tid    = threadIdx.x;
    const int waveId = tid >> 5;
    const int lane   = tid & 31;
    const int half   = (lane >> 4) & 1;
    const int r      = lane & 15;
    const int wm0    = (waveId >> 1) * 32;
    const int wn0    = (waveId & 1) * 32;
    const int gRow0  = blockIdx.x * 128;
    const int gCol0  = blockIdx.y * 64;

    // staging thread mapping
    const int ar  = tid >> 4;             // A base row (0..15), rows ar+16*i
    const int ac  = (tid & 15) * 2;       // A even col pair
    const int bn  = tid & 63;             // B column (n)
    const int bk0 = (tid >> 6) * 2;       // B even k pair base, pairs bk0+8*i

    float aF[16];
    float bF[8];

    auto load_tile = [&](int kb) {
        #pragma unroll
        for (int i = 0; i < 8; ++i) {
            const int gr = gRow0 + ar + i * 16;
            const int kk = kb + ac;
            float vx = 0.0f, vy = 0.0f;
            if (!GUARDK || kk + 1 < K) {
                if (AMODE == 0) {
                    const float2 v = *(const float2*)(A + (size_t)gr * K + kk);
                    vx = v.x; vy = v.y;
                } else {
                    const int c = kk % Cx;
                    const float2 xv = *(const float2*)(X + (size_t)gr * Cx + c);
                    const float2 sv = *(const float2*)(S + (size_t)(gr % NDIM) * K + kk);
                    vx = xv.x * sv.x; vy = xv.y * sv.y;
                }
            } else if (GUARDK && kk < K) {
                if (AMODE == 0) vx = A[(size_t)gr * K + kk];
                else vx = X[(size_t)gr * Cx + (kk % Cx)] *
                          S[(size_t)(gr % NDIM) * K + kk];
            }
            aF[2 * i] = vx; aF[2 * i + 1] = vy;
        }
        #pragma unroll
        for (int i = 0; i < 4; ++i) {
            const int kk = kb + bk0 + i * 8;
            float v0 = 0.0f, v1 = 0.0f;
            if (!GUARDK || kk < K)     v0 = Bw[(size_t)kk * Nn + gCol0 + bn];
            if (!GUARDK || kk + 1 < K) v1 = Bw[(size_t)(kk + 1) * Nn + gCol0 + bn];
            bF[2 * i] = v0; bF[2 * i + 1] = v1;
        }
    };
    auto store_tile = [&]() {
        #pragma unroll
        for (int i = 0; i < 8; ++i) {
            v2bf p; p[0] = (__bf16)aF[2 * i]; p[1] = (__bf16)aF[2 * i + 1];
            *(v2bf*)(As + (ar + i * 16) * 32 + ac) = p;
        }
        #pragma unroll
        for (int i = 0; i < 4; ++i) {
            v2bf p; p[0] = (__bf16)bF[2 * i]; p[1] = (__bf16)bF[2 * i + 1];
            *(v2bf*)(Bst + bn * BPAD + bk0 + i * 8) = p;
        }
    };

    v8f acc[2][2];
    #pragma unroll
    for (int a = 0; a < 2; ++a)
        #pragma unroll
        for (int b = 0; b < 2; ++b) acc[a][b] = (v8f){0,0,0,0,0,0,0,0};

    load_tile(0);
    for (int kb = 0; kb < K; kb += 32) {
        __syncthreads();
        store_tile();
        __syncthreads();

        if (kb + 32 < K) {
            if (AMODE == 0 && !GUARDK && kb + 64 < K) {
                __builtin_prefetch(A  + (size_t)(gRow0 + ar) * K + kb + 64 + ac, 0, 1);
                __builtin_prefetch(Bw + (size_t)(kb + 64 + bk0) * Nn + gCol0 + bn, 0, 1);
            }
            load_tile(kb + 32);
        }

        // ---- fragments: two contiguous 16B chunks per lane (ISA 7.12.2) ----
        v16bf af[2], bf[2];
        #pragma unroll
        for (int tm = 0; tm < 2; ++tm) {
            const int m = wm0 + tm * 16 + r;
            const v8bf lo = *(const v8bf*)(As + m * 32 + 8 * half);
            const v8bf hi = *(const v8bf*)(As + m * 32 + 16 + 8 * half);
            af[tm] = __builtin_shufflevector(lo, hi,
                0,1,2,3,4,5,6,7,8,9,10,11,12,13,14,15);
        }
        #pragma unroll
        for (int tn = 0; tn < 2; ++tn) {
            const int n = wn0 + tn * 16 + r;
            const v8bf lo = *(const v8bf*)(Bst + n * BPAD + 8 * half);
            const v8bf hi = *(const v8bf*)(Bst + n * BPAD + 16 + 8 * half);
            bf[tn] = __builtin_shufflevector(lo, hi,
                0,1,2,3,4,5,6,7,8,9,10,11,12,13,14,15);
        }
        #pragma unroll
        for (int tm = 0; tm < 2; ++tm)
            #pragma unroll
            for (int tn = 0; tn < 2; ++tn)
                acc[tm][tn] = __builtin_amdgcn_wmma_f32_16x16x32_bf16(
                    false, af[tm], false, bf[tn], (short)0, acc[tm][tn], false, false);
    }

    // ---- epilogue ----
    #pragma unroll
    for (int tm = 0; tm < 2; ++tm) {
        #pragma unroll
        for (int tn = 0; tn < 2; ++tn) {
            const int col = gCol0 + wn0 + tn * 16 + r;
            #pragma unroll
            for (int v = 0; v < 8; ++v) {
                const int row = gRow0 + wm0 + tm * 16 + half * 8 + v;
                float x = acc[tm][tn][v];
                if (EPI >= 1) x += bias[col];
                if (EPI == 2) x = gelu_f(x);
                C[(size_t)row * Nn + col] = x;
            }
        }
    }
}

// ---------------------------------------------------------------------------
// S[n, k*C+c] = rowsum(normalize(A + tanh(adj)*0.3))[k,n] * dw[k,c]
// ---------------------------------------------------------------------------
__global__ __launch_bounds__(256) void aeff_kernel(
    const float* __restrict__ A, const float* __restrict__ adj,
    const float* __restrict__ dw, float* __restrict__ S, int C)
{
    __shared__ float rs[3 * NDIM];
    const int tid = threadIdx.x;
    if (tid < 3 * NDIM) {
        const int k = tid / NDIM, n = tid % NDIM;
        const float* a  = A   + (size_t)(k * NDIM + n) * NDIM;
        const float* ar = adj + (size_t)(k * NDIM + n) * NDIM;
        float s = 0.0f, sa = 0.0f;
        for (int m = 0; m < NDIM; ++m) {
            const float v = a[m] + tanhf(ar[m]) * 0.3f;
            s += v; sa += fabsf(v);
        }
        rs[tid] = s / fmaxf(sa, 1.0f);
    }
    __syncthreads();
    const int K = 3 * C, total = NDIM * K;
    for (int idx = tid; idx < total; idx += 256) {
        const int n = idx / K, kk = idx % K, k = kk / C;
        S[idx] = rs[k * NDIM + n] * dw[kk];
    }
}

// ---------------------------------------------------------------------------
// Stage 1: LN(16) -> @proj_w+b (16->96) -> LN(96) -> GELU.  One wave per row.
// ---------------------------------------------------------------------------
__global__ __launch_bounds__(256) void ln_proj_kernel(
    const float* __restrict__ X,
    const float* __restrict__ ing, const float* __restrict__ inb,
    const float* __restrict__ W,  const float* __restrict__ pb,
    const float* __restrict__ pg, const float* __restrict__ pbeta,
    float* __restrict__ out, int rows)
{
    const int wid = threadIdx.x >> 5, lane = threadIdx.x & 31;
    const int row = blockIdx.x * 8 + wid;
    if (row >= rows) return;
    const float* xr = X + (size_t)row * 16;
    float x[16];
    float m = 0.0f;
    #pragma unroll
    for (int i = 0; i < 16; ++i) { x[i] = xr[i]; m += x[i]; }
    m *= (1.0f / 16.0f);
    float v = 0.0f;
    #pragma unroll
    for (int i = 0; i < 16; ++i) { const float d = x[i] - m; v += d * d; }
    v *= (1.0f / 16.0f);
    float rstd = rsqrtf(v + 1e-5f);
    float xl[16];
    #pragma unroll
    for (int i = 0; i < 16; ++i) xl[i] = (x[i] - m) * rstd * ing[i] + inb[i];

    float y[3];
    #pragma unroll
    for (int jj = 0; jj < 3; ++jj) {
        const int j = lane + jj * 32;
        float acc = pb[j];
        #pragma unroll
        for (int i = 0; i < 16; ++i) acc += xl[i] * W[i * 96 + j];
        y[jj] = acc;
    }
    float s = y[0] + y[1] + y[2];
    float ss = y[0]*y[0] + y[1]*y[1] + y[2]*y[2];
    s = wave_sum(s); ss = wave_sum(ss);
    const float mean = s * (1.0f / 96.0f);
    const float var  = ss * (1.0f / 96.0f) - mean * mean;
    rstd = rsqrtf(var + 1e-5f);
    #pragma unroll
    for (int jj = 0; jj < 3; ++jj) {
        const int j = lane + jj * 32;
        const float t = (y[jj] - mean) * rstd * pg[j] + pbeta[j];
        out[(size_t)row * 96 + j] = gelu_f(t);
    }
}

// ---------------------------------------------------------------------------
// Generic wave-per-row LayerNorm: out = LN(in (+res)) * g + b, opt GELU.
// D must be a multiple of 32 and <= 256.
// ---------------------------------------------------------------------------
__global__ __launch_bounds__(256) void ln_rows_kernel(
    const float* __restrict__ in, const float* __restrict__ res,
    const float* __restrict__ g,  const float* __restrict__ b,
    float* __restrict__ out, int D, int rows, int doGelu)
{
    const int wid = threadIdx.x >> 5, lane = threadIdx.x & 31;
    const int row = blockIdx.x * 8 + wid;
    if (row >= rows) return;
    const int per = D >> 5;
    float v[8];
    float s = 0.0f, ss = 0.0f;
    #pragma unroll 8
    for (int i = 0; i < per; ++i) {
        const int c = lane + i * 32;
        float t = in[(size_t)row * D + c];
        if (res) t += res[(size_t)row * D + c];
        v[i] = t; s += t; ss += t * t;
    }
    s = wave_sum(s); ss = wave_sum(ss);
    const float mean = s / (float)D;
    const float var  = ss / (float)D - mean * mean;
    const float rstd = rsqrtf(var + 1e-5f);
    #pragma unroll 8
    for (int i = 0; i < per; ++i) {
        const int c = lane + i * 32;
        float t = (v[i] - mean) * rstd * g[c] + b[c];
        if (doGelu) t = gelu_f(t);
        out[(size_t)row * D + c] = t;
    }
}

// ---------------------------------------------------------------------------
// Depthwise temporal conv (kernel k, pad k/2) + GroupNorm(8) fused.
// One block per (b, n, group).  Input/output: [B,T,N,Co] row-major.
// ---------------------------------------------------------------------------
__global__ __launch_bounds__(256) void dwconv_gn_kernel(
    const float* __restrict__ hin, const float* __restrict__ w,
    const float* __restrict__ gg,  const float* __restrict__ gb,
    float* __restrict__ out, int Co, int ksz)
{
    const int Cg  = Co >> 3;
    const int blk = blockIdx.x;
    const int g   = blk & 7;
    const int n   = (blk >> 3) % NDIM;
    const int bb  = blk / (NDIM * 8);
    const int tid = threadIdx.x;
    const int cnt = Cg * TDIM;             // 768 or 1024
    const int E   = (cnt + 255) >> 8;      // 3 or 4

    float vals[4];
    float s = 0.0f, ss = 0.0f;
    for (int e = 0; e < E; ++e) {
        const int idx = tid + (e << 8);
        float acc = 0.0f;
        if (idx < cnt) {
            const int cl = idx >> 5;       // channel-in-group
            const int t  = idx & 31;
            const int co = g * Cg + cl;
            for (int j = 0; j < ksz; ++j) {
                const int tt = t + j - (ksz >> 1);
                if (tt >= 0 && tt < TDIM)
                    acc += w[co * ksz + j] *
                           hin[((size_t)(bb * TDIM + tt) * NDIM + n) * Co + co];
            }
        }
        vals[e] = acc; s += acc; ss += acc * acc;
    }
    __shared__ float rs_[256], rss_[256];
    rs_[tid] = s; rss_[tid] = ss;
    __syncthreads();
    for (int o = 128; o >= 1; o >>= 1) {
        if (tid < o) { rs_[tid] += rs_[tid + o]; rss_[tid] += rss_[tid + o]; }
        __syncthreads();
    }
    const float mean = rs_[0] / (float)cnt;
    const float var  = rss_[0] / (float)cnt - mean * mean;
    const float rstd = rsqrtf(var + 1e-5f);
    for (int e = 0; e < E; ++e) {
        const int idx = tid + (e << 8);
        if (idx < cnt) {
            const int cl = idx >> 5;
            const int t  = idx & 31;
            const int co = g * Cg + cl;
            out[((size_t)(bb * TDIM + t) * NDIM + n) * Co + co] =
                (vals[e] - mean) * rstd * gg[co] + gb[co];
        }
    }
}

// ---------------------------------------------------------------------------
// Node-attention pooling: s = sbuf @ w2 + b2; softmax over N=47; pooled.
// One block per (b,t).
// ---------------------------------------------------------------------------
__global__ __launch_bounds__(256) void pool_kernel(
    const float* __restrict__ x, const float* __restrict__ sbuf,
    const float* __restrict__ w2, const float* __restrict__ b2,
    float* __restrict__ pooled)
{
    const int bt = blockIdx.x, tid = threadIdx.x;
    __shared__ float sw[NDIM];
    __shared__ float w2s[64];
    if (tid < 64) w2s[tid] = w2[tid];
    __syncthreads();
    if (tid < NDIM) {
        const float* sr = sbuf + ((size_t)bt * NDIM + tid) * 64;
        float acc = b2[0];
        for (int i = 0; i < 64; ++i) acc += sr[i] * w2s[i];
        sw[tid] = acc;
    }
    __syncthreads();
    if (tid == 0) {
        float mx = -1e30f;
        for (int n = 0; n < NDIM; ++n) mx = fmaxf(mx, sw[n]);
        float sum = 0.0f;
        for (int n = 0; n < NDIM; ++n) { const float e = expf(sw[n] - mx); sw[n] = e; sum += e; }
        const float inv = 1.0f / sum;
        for (int n = 0; n < NDIM; ++n) sw[n] *= inv;
    }
    __syncthreads();
    float acc = 0.0f;
    for (int n = 0; n < NDIM; ++n)
        acc += sw[n] * x[((size_t)bt * NDIM + n) * DDIM + tid];
    pooled[(size_t)bt * DDIM + tid] = acc;
}

// ---------------------------------------------------------------------------
// cbuf[bt] = concat(pooled[bt], LN76(geo[bt]))   (332 wide)
// ---------------------------------------------------------------------------
__global__ __launch_bounds__(256) void concat_geo_kernel(
    const float* __restrict__ pooled, const float* __restrict__ geo,
    const float* __restrict__ gg, const float* __restrict__ gb,
    float* __restrict__ cbuf)
{
    const int bt = blockIdx.x, tid = threadIdx.x;
    cbuf[(size_t)bt * 332 + tid] = pooled[(size_t)bt * DDIM + tid];
    if (tid < 32) {
        float v[3];
        float s = 0.0f, ss = 0.0f;
        #pragma unroll
        for (int k = 0; k < 3; ++k) {
            const int c = tid + k * 32;
            const float t = (c < 76) ? geo[(size_t)bt * 76 + c] : 0.0f;
            v[k] = t; s += t; ss += t * t;
        }
        s = wave_sum(s); ss = wave_sum(ss);
        const float mean = s * (1.0f / 76.0f);
        const float var  = ss * (1.0f / 76.0f) - mean * mean;
        const float rstd = rsqrtf(var + 1e-5f);
        #pragma unroll
        for (int k = 0; k < 3; ++k) {
            const int c = tid + k * 32;
            if (c < 76)
                cbuf[(size_t)bt * 332 + 256 + c] = (v[k] - mean) * rstd * gg[c] + gb[c];
        }
    }
}

__global__ __launch_bounds__(256) void add_pos_kernel(
    float* __restrict__ tok, const float* __restrict__ pos)
{
    const int bt = blockIdx.x, d = threadIdx.x;
    tok[(size_t)bt * DDIM + d] += pos[(bt % TDIM) * DDIM + d];
}

__global__ __launch_bounds__(256) void add_inplace_kernel(
    float* __restrict__ y, const float* __restrict__ x, int n)
{
    const int i = blockIdx.x * 256 + threadIdx.x;
    if (i < n) y[i] += x[i];
}

// ---------------------------------------------------------------------------
// MHA core: per (b, head), T=32, hd=32.  Block (32,32): wave = query row.
// ---------------------------------------------------------------------------
__global__ __launch_bounds__(1024) void attn_kernel(
    const float* __restrict__ qkv, float* __restrict__ obuf)
{
    const int bh = blockIdx.x;
    const int b  = bh >> 3;
    const int h  = bh & 7;
    const int j  = threadIdx.x;   // lane
    const int i  = threadIdx.y;   // wave == query row t
    __shared__ float qs[32][33], ks[32][33], vs[32][33];
    const size_t base = (size_t)(b * 32 + i) * 768 + h * 32 + j;
    qs[i][j] = qkv[base];
    ks[i][j] = qkv[base + 256];
    vs[i][j] = qkv[base + 512];
    __syncthreads();
    float acc = 0.0f;
    #pragma unroll
    for (int d = 0; d < 32; ++d) acc += qs[i][d] * ks[j][d];
    acc *= 0.17677669529663687f;          // 1/sqrt(32)
    const float mx = wave_max(acc);
    const float e  = expf(acc - mx);
    const float s  = wave_sum(e);
    const float att = e / s;
    float out = 0.0f;
    #pragma unroll
    for (int jj = 0; jj < 32; ++jj) {
        const float a = __shfl(att, jj, 32);
        out += a * vs[jj][j];
    }
    obuf[(size_t)(b * 32 + i) * DDIM + h * 32 + j] = out;
}

// ---------------------------------------------------------------------------
// Host-side orchestration
// ---------------------------------------------------------------------------
// Input leaf indices (depth-first, dict insertion order, None skipped):
enum {
    IDX_X = 0, IDX_GEO = 1,
    IN_LN_G = 2, IN_LN_B = 3, PROJ_W = 4, PROJ_B = 5, PROJ_LN_G = 6, PROJ_LN_B = 7,
    GCN0 = 8,   // dw,pw_w,tconv_w,gn_g,gn_b,ln_g,ln_b,adj_res,res_w  (9 leaves)
    GCN1 = 17,  // dw..adj_res (8 leaves, no res_w)
    GCN2 = 25,  // 9 leaves
    NA_W1 = 34, NA_B1 = 35, NA_W2 = 36, NA_B2 = 37,
    GEO_LN_G = 38, GEO_LN_B = 39, GEO_W = 40, GEO_B = 41, POS = 42,
    LAYER0 = 43,  // per layer: ln1_g,ln1_b,in_w,in_b,out_w,out_b,ln2_g,ln2_b,ff1_w,ff1_b,ff2_w,ff2_b
    FIN_G = 91, FIN_B = 92, IDX_A = 93
};

extern "C" void kernel_launch(void* const* d_in, const int* in_sizes, int n_in,
                              void* d_out, int out_size, void* d_ws, size_t ws_size,
                              hipStream_t stream)
{
    (void)in_sizes; (void)n_in; (void)out_size; (void)ws_size;
    auto in = [&](int i) { return (const float*)d_in[i]; };

    float* ws = (float*)d_ws;
    const size_t BIG = (size_t)M1 * 256;
    float* S      = ws;                    // 47*576 <= 32768
    float* bufA   = ws + 32768;            // activations  [M1,<=256]
    float* bufB   = bufA + BIG;            // residual
    float* bufC   = bufB + BIG;            // gemm out / na scores
    float* bufD   = bufC + BIG;            // conv+GN out
    float* pooled = bufD + BIG;            // [M2,256]
    float* cbuf   = pooled + (size_t)M2 * 256;   // [M2,332]
    float* tok    = cbuf + (size_t)M2 * 332;     // [M2,256]
    float* lbuf   = tok + (size_t)M2 * 256;      // [M2,256]
    float* qkv    = lbuf + (size_t)M2 * 256;     // [M2,768]
    float* obuf   = qkv + (size_t)M2 * 768;      // [M2,256]
    float* fbuf   = obuf + (size_t)M2 * 256;     // [M2,1024]
    float* pbuf   = fbuf + (size_t)M2 * 1024;    // [M2,256]

    auto gemm = [&](const float* A, const float* B, const float* bias, float* C,
                    int M, int N, int K, int epi, bool guard) {
        dim3 g(M / 128, N / 64);
        if (guard) {
            gemm_kernel<0,1,true><<<g, 256, 0, stream>>>(A, B, bias, C, M, N, K, nullptr, nullptr, 0);
        } else if (epi == 0) {
            gemm_kernel<0,0,false><<<g, 256, 0, stream>>>(A, B, bias, C, M, N, K, nullptr, nullptr, 0);
        } else if (epi == 1) {
            gemm_kernel<0,1,false><<<g, 256, 0, stream>>>(A, B, bias, C, M, N, K, nullptr, nullptr, 0);
        } else {
            gemm_kernel<0,2,false><<<g, 256, 0, stream>>>(A, B, bias, C, M, N, K, nullptr, nullptr, 0);
        }
    };

    // ---- stage 1: LN(16) -> proj(16->96) -> LN -> GELU ----
    ln_proj_kernel<<<M1 / 8, 256, 0, stream>>>(
        in(IDX_X), in(IN_LN_G), in(IN_LN_B), in(PROJ_W), in(PROJ_B),
        in(PROJ_LN_G), in(PROJ_LN_B), bufA, M1);

    // ---- three DSGCN blocks ----
    const int base[3] = {GCN0, GCN1, GCN2};
    const int ci[3]   = {96, 192, 192};
    const int co[3]   = {192, 192, 256};
    const int kk_[3]  = {3, 3, 5};
    const bool hasRes[3] = {true, false, true};

    for (int blk = 0; blk < 3; ++blk) {
        const int b0 = base[blk];
        const float* dw    = in(b0 + 0);
        const float* pw    = in(b0 + 1);
        const float* tconv = in(b0 + 2);
        const float* gn_g  = in(b0 + 3);
        const float* gn_b  = in(b0 + 4);
        const float* ln_g  = in(b0 + 5);
        const float* ln_b  = in(b0 + 6);
        const float* adj   = in(b0 + 7);

        aeff_kernel<<<1, 256, 0, stream>>>(in(IDX_A), adj, dw, S, ci[blk]);

        const float* resPtr;
        if (hasRes[blk]) {
            gemm(bufA, in(b0 + 8), nullptr, bufB, M1, co[blk], ci[blk], 0, false);
            resPtr = bufB;
        } else {
            resPtr = bufA;
        }

        // GCN expand GEMM via scale table (AMODE=1)
        {
            dim3 g(M1 / 128, co[blk] / 64);
            gemm_kernel<1,0,false><<<g, 256, 0, stream>>>(
                nullptr, pw, nullptr, bufC, M1, co[blk], 3 * ci[blk], bufA, S, ci[blk]);
        }

        dwconv_gn_kernel<<<BDIM * NDIM * 8, 256, 0, stream>>>(
            bufC, tconv, gn_g, gn_b, bufD, co[blk], kk_[blk]);

        ln_rows_kernel<<<M1 / 8, 256, 0, stream>>>(
            bufD, resPtr, ln_g, ln_b, bufA, co[blk], M1, 1);
    }

    // ---- node-attention pooling ----
    gemm(bufA, in(NA_W1), in(NA_B1), bufC, M1, 64, 256, 2, false);   // GELU(x@w1+b1)
    pool_kernel<<<M2, 256, 0, stream>>>(bufA, bufC, in(NA_W2), in(NA_B2), pooled);

    // ---- geo fusion ----
    concat_geo_kernel<<<M2, 256, 0, stream>>>(pooled, in(IDX_GEO),
                                              in(GEO_LN_G), in(GEO_LN_B), cbuf);
    gemm(cbuf, in(GEO_W), in(GEO_B), tok, M2, 256, 332, 1, true);
    add_pos_kernel<<<M2, 256, 0, stream>>>(tok, in(POS));

    // ---- 4 transformer layers (norm-first) ----
    for (int l = 0; l < 4; ++l) {
        const int lb = LAYER0 + 12 * l;
        const float* ln1_g = in(lb + 0); const float* ln1_b = in(lb + 1);
        const float* in_w  = in(lb + 2); const float* in_b  = in(lb + 3);
        const float* out_w = in(lb + 4); const float* out_b = in(lb + 5);
        const float* ln2_g = in(lb + 6); const float* ln2_b = in(lb + 7);
        const float* ff1_w = in(lb + 8); const float* ff1_b = in(lb + 9);
        const float* ff2_w = in(lb + 10); const float* ff2_b = in(lb + 11);

        ln_rows_kernel<<<M2 / 8, 256, 0, stream>>>(tok, nullptr, ln1_g, ln1_b, lbuf, 256, M2, 0);
        gemm(lbuf, in_w, in_b, qkv, M2, 768, 256, 1, false);
        attn_kernel<<<BDIM * 8, dim3(32, 32), 0, stream>>>(qkv, obuf);
        gemm(obuf, out_w, out_b, pbuf, M2, 256, 256, 1, false);
        add_inplace_kernel<<<(M2 * 256) / 256, 256, 0, stream>>>(tok, pbuf, M2 * 256);

        ln_rows_kernel<<<M2 / 8, 256, 0, stream>>>(tok, nullptr, ln2_g, ln2_b, lbuf, 256, M2, 0);
        gemm(lbuf, ff1_w, ff1_b, fbuf, M2, 1024, 256, 2, false);
        gemm(fbuf, ff2_w, ff2_b, pbuf, M2, 256, 1024, 1, false);
        add_inplace_kernel<<<(M2 * 256) / 256, 256, 0, stream>>>(tok, pbuf, M2 * 256);
    }

    // ---- final LN -> output ----
    ln_rows_kernel<<<M2 / 8, 256, 0, stream>>>(tok, nullptr, in(FIN_G), in(FIN_B),
                                               (float*)d_out, 256, M2, 0);
}
